// DynamicWaveAttention_5995774345692
// MI455X (gfx1250) — compile-verified, matrix operand order checked by reading, not run on hardware
//
#include <hip/hip_runtime.h>

#define S_LEN    2048
#define DM       512
#define NH       8
#define HD       64
#define GH       128
#define WF       16
#define BATCH    2
#define BS       (BATCH * S_LEN)      // 4096 tokens
#define LOG2E    1.4426950408889634f

typedef __attribute__((ext_vector_type(16))) __bf16 v16bf;
typedef __attribute__((ext_vector_type(8)))  __bf16 v8bf;
typedef __attribute__((ext_vector_type(8)))  float  v8f;

// D = A(16x32 bf16) * B(32x16 bf16) + C(16x16 f32), wave32.
static __device__ __forceinline__ v8f wmma_bf(v16bf a, v16bf b, v8f c) {
  return __builtin_amdgcn_wmma_f32_16x16x32_bf16(false, a, false, b, (short)0, c,
                                                 false, false);
}

// A/B fragment loader: 16-bit 16x32 layout (ISA 7.12.2):
//   lanes 0-15 : K = 0..7 (elems 0-7), 16..23 (elems 8-15)
//   lanes 16-31: K = 8..15,            24..31
// -> two contiguous 16B loads per lane from a row-major row pointer.
static __device__ __forceinline__ v16bf load_frag(const __bf16* row, int hi) {
  v8bf lo = *(const v8bf*)(row + hi);
  v8bf hh = *(const v8bf*)(row + hi + 16);
  v16bf r;
#pragma unroll
  for (int i = 0; i < 8; ++i) { r[i] = lo[i]; r[i + 8] = hh[i]; }
  return r;
}

// ---------------------------------------------------------------------------
// 0) fp32 -> bf16 conversion of x, and bf16 transpose of Wq/Wk/Wv ([in,out] ->
//    Wt[out,in]) so GEMM B-fragments are contiguous loads.
// ---------------------------------------------------------------------------
__global__ __launch_bounds__(256) void prep_kernel(
    const float* __restrict__ x, const float* __restrict__ wq,
    const float* __restrict__ wk, const float* __restrict__ wv,
    __bf16* __restrict__ xb, __bf16* __restrict__ wtb) {
  size_t i = (size_t)blockIdx.x * blockDim.x + threadIdx.x;
  const size_t NX = (size_t)BS * DM;       // 2,097,152
  const size_t NW = (size_t)DM * DM;       // 262,144
  if (i < NX) xb[i] = (__bf16)x[i];
  if (i < 3 * NW) {
    size_t m = i / NW, r = i % NW;         // r = o*DM + in
    size_t o = r >> 9, in = r & (DM - 1);
    const float* w = (m == 0) ? wq : (m == 1) ? wk : wv;
    wtb[i] = (__bf16)w[in * DM + o];
  }
}

// ---------------------------------------------------------------------------
// 1) gate MLP: gates[b,h,s] = sigmoid(relu(x@g1+b1)@g2+b2)   (head-major so
//    the attention kernel reads 8 contiguous floats per lane per tile)
// ---------------------------------------------------------------------------
__global__ __launch_bounds__(128) void gate_kernel(
    const float* __restrict__ x, const float* __restrict__ g1w,
    const float* __restrict__ g1b, const float* __restrict__ g2w,
    const float* __restrict__ g2b, float* __restrict__ gates) {
  __shared__ float xs[DM];
  __shared__ float h1[GH];
  const int tok = blockIdx.x;              // 0..4095
  const int t = threadIdx.x;               // 0..127
  for (int i = t; i < DM; i += 128) xs[i] = x[(size_t)tok * DM + i];
  __syncthreads();
  float acc = g1b[t];
#pragma unroll 8
  for (int k = 0; k < DM; ++k) acc = fmaf(xs[k], g1w[k * GH + t], acc);
  h1[t] = fmaxf(acc, 0.0f);
  __syncthreads();
  if (t < NH) {
    float a = g2b[t];
#pragma unroll 8
    for (int j = 0; j < GH; ++j) a = fmaf(h1[j], g2w[j * NH + t], a);
    float sg = 1.0f / (1.0f + exp2f(-a * LOG2E));
    int b = tok >> 11, s = tok & (S_LEN - 1);
    gates[((size_t)(b * NH + t)) * S_LEN + s] = sg;
  }
}

// ---------------------------------------------------------------------------
// 2) QKV projection: one wave computes one 16x16 tile, K=512 via 16 chained
//    bf16 WMMAs.  grid = (BS/16, DM/64, 3), block = 128 (4 waves).
//    Q,K stored [b,h,s,d] bf16; V stored transposed [b,h,d,s] bf16 so the
//    attention O^T GEMM gets contiguous A-fragments.
// ---------------------------------------------------------------------------
__global__ __launch_bounds__(128) void qkv_gemm_kernel(
    const __bf16* __restrict__ xb, const __bf16* __restrict__ wtb,
    const float* __restrict__ bq, const float* __restrict__ bk,
    const float* __restrict__ bv, __bf16* __restrict__ Qb,
    __bf16* __restrict__ Kb, __bf16* __restrict__ VTb) {
  const int tt   = blockIdx.x;                  // token 16-tile
  const int mat  = blockIdx.z;                  // 0=Q 1=K 2=V
  const int wave = threadIdx.x >> 5;
  const int lane = threadIdx.x & 31;
  const int l16  = lane & 15;
  const int hi   = (lane >> 4) << 3;            // 0 or 8
  const int obase = blockIdx.y * 64 + wave * 16;

  const __bf16* arow = xb + (size_t)(tt * 16 + l16) * DM;
  const __bf16* brow = wtb + ((size_t)mat * DM + obase + l16) * DM;

  v8f c = {};
#pragma unroll 4
  for (int kk = 0; kk < DM; kk += 32) {
    v16bf a = load_frag(arow + kk, hi);
    v16bf b = load_frag(brow + kk, hi);
    c = wmma_bf(a, b, c);
  }

  const int o = obase + l16, h = o >> 6, d = o & (HD - 1);
  const int tok0 = tt * 16, b = tok0 >> 11, s0 = tok0 & (S_LEN - 1);
  const float* bias = (mat == 0) ? bq : (mat == 1) ? bk : bv;
  const float bvv = bias[o];
#pragma unroll
  for (int r = 0; r < 8; ++r) c[r] += bvv;

  if (mat == 2) {                              // V^T: 8 consecutive s -> one 16B store
    v8bf vv;
#pragma unroll
    for (int r = 0; r < 8; ++r) vv[r] = (__bf16)c[r];
    __bf16* p = VTb + (((size_t)(b * NH + h) * HD + d) * S_LEN + s0 + hi);
    *(v8bf*)p = vv;
  } else {
    __bf16* base = ((mat == 0) ? Qb : Kb) +
                   (((size_t)(b * NH + h) * S_LEN + s0 + hi) * HD + d);
#pragma unroll
    for (int r = 0; r < 8; ++r) base[(size_t)r * HD] = (__bf16)c[r];
  }
}

// ---------------------------------------------------------------------------
// 3) fused flash attention (transposed): per (b, 16-query tile) workgroup of
//    8 waves (one head each).  S^T = K*Q^T  ->  affine (gate+wavesim, adj)
//    ->  online softmax (lane-local + 1 shfl_xor)  ->  O^T += V^T * P^T.
//    64-key staging: adj tile pulled straight into LDS with CDNA5 async-LDS
//    DMA (global_load_async_to_lds_b128, ASYNCcnt), wave-sim tile computed by
//    VALU in the shadow of the DMA; both reused by all 8 heads.
// ---------------------------------------------------------------------------
__global__ __launch_bounds__(256) void attn_kernel(
    const __bf16* __restrict__ Qb, const __bf16* __restrict__ Kb,
    const __bf16* __restrict__ VTb, const float* __restrict__ gates,
    const float* __restrict__ wmask, const float* __restrict__ adj,
    float* __restrict__ OT) {
  __shared__ float wmq[16][WF];         // wave_mask rows of the q tile
  __shared__ float ws_lds[16][64];      // wave_sim for current 64-key chunk
  __shared__ float adj_lds[16][64];     // adj tile (filled by async-LDS DMA)

  const int b     = blockIdx.x >> 7;    // S/16 = 128 q tiles
  const int qt    = blockIdx.x & 127;
  const int qbase = qt * 16;
  const int h     = threadIdx.x >> 5;   // wave == head
  const int lane  = threadIdx.x & 31;
  const int l16   = lane & 15;
  const int hi    = (lane >> 4) << 3;   // 0 or 8

  // stage the 16 query wave-mask rows (256 threads -> 1 elem each)
  wmq[threadIdx.x >> 4][threadIdx.x & 15] =
      wmask[((size_t)b * S_LEN + qbase + (threadIdx.x >> 4)) * WF + (threadIdx.x & 15)];
  __syncthreads();

  // Q^T B-fragments (column = this lane's query, contiguous head-dim elems)
  const __bf16* qrow = Qb + ((size_t)(b * NH + h) * S_LEN + qbase + l16) * HD;
  const v16bf qf0 = load_frag(qrow, hi);        // d 0..31
  const v16bf qf1 = load_frag(qrow + 32, hi);   // d 32..63

  v8f oacc[4] = {v8f{}, v8f{}, v8f{}, v8f{}};
  float m = -3.0e38f, l = 0.0f;

  const float* gp_base = gates + (size_t)(b * NH + h) * S_LEN;
  const __bf16* krow_base = Kb + ((size_t)(b * NH + h) * S_LEN + l16) * HD;

  // addresses for this thread's async 16B slice of the adj tile (q, 4 keys)
  const int sq  = threadIdx.x >> 4;             // 0..15
  const int sk4 = (threadIdx.x & 15) * 4;       // 0..60
  const float* adj_row = adj + ((size_t)b * S_LEN + qbase + sq) * S_LEN + sk4;
  const unsigned adj_lds_off = (unsigned)(uintptr_t)&adj_lds[sq][sk4];

  for (int kc = 0; kc < S_LEN / 64; ++kc) {
    const int kbase64 = kc * 64;

    // ---- async DMA: adj 16x64 tile -> LDS, one b128 per thread ----
    {
      unsigned long long ga = (unsigned long long)(uintptr_t)(adj_row + kbase64);
      asm volatile("global_load_async_to_lds_b128 %0, %1, off"
                   :: "v"(adj_lds_off), "v"(ga) : "memory");
    }
    // ---- wave-sim tile on VALU in the DMA shadow (4 cells per thread) ----
#pragma unroll
    for (int i = 0; i < 4; ++i) {
      int cell = threadIdx.x * 4 + i;           // 0..1023
      int q = cell >> 6, kk = cell & 63;
      const float* wmk = wmask + ((size_t)b * S_LEN + kbase64 + kk) * WF;
      float acc = 0.0f;
#pragma unroll
      for (int f = 0; f < WF; ++f) acc = fmaf(wmq[q][f], wmk[f], acc);
      ws_lds[q][kk] = acc;
    }
    asm volatile("s_wait_asynccnt 0x0" ::: "memory");
    __syncthreads();

    if (kc + 1 < S_LEN / 64) {  // prefetch next chunk (global_prefetch_b8)
      __builtin_prefetch(krow_base + (size_t)(kbase64 + 64) * HD, 0, 0);
      __builtin_prefetch(VTb + ((size_t)(b * NH + h) * HD + l16) * S_LEN + kbase64 + 64, 0, 0);
    }

    // ---- two 32-key halves of the staged chunk ----
#pragma unroll
    for (int j = 0; j < 2; ++j) {
      const int kbase = kbase64 + j * 32;
      const int ko = j * 32;                    // offset into LDS tiles

      // S^T = K * Q^T : two 16x16 key tiles, K-dim = 64 (2 WMMAs each)
      const __bf16* krow = krow_base + (size_t)kbase * HD;
      v8f s0 = {}, s1 = {};
      {
        v16bf ka = load_frag(krow, hi);
        v16bf kb = load_frag(krow + 32, hi);
        s0 = wmma_bf(ka, qf0, s0);
        s0 = wmma_bf(kb, qf1, s0);
        ka = load_frag(krow + 16 * HD, hi);
        kb = load_frag(krow + 16 * HD + 32, hi);
        s1 = wmma_bf(ka, qf0, s1);
        s1 = wmma_bf(kb, qf1, s1);
      }

      // per-key gates (contiguous, head-major layout)
      const float* gp = gp_base + kbase + hi;
      float4 ga = ((const float4*)gp)[0], gb = ((const float4*)gp)[1];
      float4 gc = ((const float4*)(gp + 16))[0], gd = ((const float4*)(gp + 16))[1];
      float gA[8] = {ga.x, ga.y, ga.z, ga.w, gb.x, gb.y, gb.z, gb.w};
      float gB[8] = {gc.x, gc.y, gc.z, gc.w, gd.x, gd.y, gd.z, gd.w};

      // affine + online softmax (lane holds one query column)
      float p0[8], p1[8], mloc = -3.0e38f;
#pragma unroll
      for (int r = 0; r < 8; ++r) {
        float w0 = ws_lds[l16][ko + hi + r],      a0 = adj_lds[l16][ko + hi + r];
        float w1 = ws_lds[l16][ko + 16 + hi + r], a1 = adj_lds[l16][ko + 16 + hi + r];
        float t0 = s0[r] * 0.125f * (gA[r] + w0) + a0;   // 1/sqrt(64)
        float t1 = s1[r] * 0.125f * (gB[r] + w1) + a1;
        p0[r] = t0; p1[r] = t1;
        mloc = fmaxf(mloc, fmaxf(t0, t1));
      }
      float mch  = fmaxf(mloc, __shfl_xor(mloc, 16, 32));
      float mnew = fmaxf(m, mch);
      float corr = exp2f((m - mnew) * LOG2E);
      float lsum = 0.0f;
#pragma unroll
      for (int r = 0; r < 8; ++r) {
        p0[r] = exp2f((p0[r] - mnew) * LOG2E);
        p1[r] = exp2f((p1[r] - mnew) * LOG2E);
        lsum += p0[r] + p1[r];
      }
      lsum += __shfl_xor(lsum, 16, 32);
      l = l * corr + lsum;
      m = mnew;
#pragma unroll
      for (int t = 0; t < 4; ++t)
#pragma unroll
        for (int r = 0; r < 8; ++r) oacc[t][r] *= corr;

      // P^T C-layout == B-fragment layout: lane-local bf16 pack
      v16bf pf;
#pragma unroll
      for (int r = 0; r < 8; ++r) { pf[r] = (__bf16)p0[r]; pf[r + 8] = (__bf16)p1[r]; }

      // O^T += V^T * P^T  (4 d-tiles of 16)
#pragma unroll
      for (int t = 0; t < 4; ++t) {
        const __bf16* vrow =
            VTb + ((size_t)(b * NH + h) * HD + t * 16 + l16) * S_LEN + kbase;
        v16bf vf = load_frag(vrow, hi);
        oacc[t] = wmma_bf(vf, pf, oacc[t]);
      }
    }
    __syncthreads();   // protect LDS tiles before next chunk's staging
  }

  const float inv = 1.0f / l;
#pragma unroll
  for (int t = 0; t < 4; ++t)
#pragma unroll
    for (int r = 0; r < 8; ++r) {
      int d = t * 16 + r + hi;
      OT[((size_t)(b * NH + h) * HD + d) * S_LEN + qbase + l16] = oacc[t][r] * inv;
    }
}

// ---------------------------------------------------------------------------
// 4) residual + LayerNorm.  O^T gather is L2-resident (8 MB).
// ---------------------------------------------------------------------------
__global__ __launch_bounds__(256) void ln_kernel(
    const float* __restrict__ x, const float* __restrict__ OT,
    const float* __restrict__ lng, const float* __restrict__ lnb,
    float* __restrict__ out) {
  __shared__ float red[256];
  const int tok = blockIdx.x, b = tok >> 11, s = tok & (S_LEN - 1);
  const int t = threadIdx.x;
  float y[2], sum = 0.0f;
#pragma unroll
  for (int i = 0; i < 2; ++i) {
    int d = t + i * 256, h = d >> 6, dd = d & (HD - 1);
    float o = OT[((size_t)(b * NH + h) * HD + dd) * S_LEN + s];
    y[i] = o + x[(size_t)tok * DM + d];
    sum += y[i];
  }
  red[t] = sum; __syncthreads();
  for (int st = 128; st > 0; st >>= 1) { if (t < st) red[t] += red[t + st]; __syncthreads(); }
  float mu = red[0] * (1.0f / DM);
  __syncthreads();
  float vs = 0.0f;
#pragma unroll
  for (int i = 0; i < 2; ++i) { float d0 = y[i] - mu; vs += d0 * d0; }
  red[t] = vs; __syncthreads();
  for (int st = 128; st > 0; st >>= 1) { if (t < st) red[t] += red[t + st]; __syncthreads(); }
  float rstd = rsqrtf(red[0] * (1.0f / DM) + 1e-5f);
#pragma unroll
  for (int i = 0; i < 2; ++i) {
    int d = t + i * 256;
    out[(size_t)tok * DM + d] = (y[i] - mu) * rstd * lng[d] + lnb[d];
  }
}

// ---------------------------------------------------------------------------
extern "C" void kernel_launch(void* const* d_in, const int* in_sizes, int n_in,
                              void* d_out, int out_size, void* d_ws, size_t ws_size,
                              hipStream_t stream) {
  (void)in_sizes; (void)n_in; (void)out_size; (void)ws_size;
  const float* x     = (const float*)d_in[0];
  const float* wmask = (const float*)d_in[1];
  const float* adj   = (const float*)d_in[2];
  const float* wq_w  = (const float*)d_in[3];
  const float* wq_b  = (const float*)d_in[4];
  const float* wk_w  = (const float*)d_in[5];
  const float* wk_b  = (const float*)d_in[6];
  const float* wv_w  = (const float*)d_in[7];
  const float* wv_b  = (const float*)d_in[8];
  const float* g1_w  = (const float*)d_in[9];
  const float* g1_b  = (const float*)d_in[10];
  const float* g2_w  = (const float*)d_in[11];
  const float* g2_b  = (const float*)d_in[12];
  const float* ln_g  = (const float*)d_in[13];
  const float* ln_b  = (const float*)d_in[14];
  float* out = (float*)d_out;

  uint8_t* ws = (uint8_t*)d_ws;
  __bf16* xb    = (__bf16*)(ws);                         //  4.0 MB
  __bf16* wtb   = (__bf16*)(ws + (4u << 20));            //  1.5 MB
  __bf16* Qb    = (__bf16*)(ws + (6u << 20));            //  4.0 MB
  __bf16* Kb    = (__bf16*)(ws + (10u << 20));           //  4.0 MB
  __bf16* VTb   = (__bf16*)(ws + (14u << 20));           //  4.0 MB
  float*  gates = (float*) (ws + (18u << 20));           //  128 KB
  float*  OT    = (float*) (ws + (19u << 20));           //  8.0 MB  (total 27 MB)

  prep_kernel<<<8192, 256, 0, stream>>>(x, wq_w, wk_w, wv_w, xb, wtb);
  gate_kernel<<<BS, 128, 0, stream>>>(x, g1_w, g1_b, g2_w, g2_b, gates);
  qkv_gemm_kernel<<<dim3(BS / 16, DM / 64, 3), 128, 0, stream>>>(
      xb, wtb, wq_b, wk_b, wv_b, Qb, Kb, VTb);
  attn_kernel<<<BATCH * (S_LEN / 16), 256, 0, stream>>>(
      Qb, Kb, VTb, gates, wmask, adj, OT);
  ln_kernel<<<BS, 256, 0, stream>>>(x, OT, ln_g, ln_b, out);
}
